// SLSTMBlock_28939489640964
// MI455X (gfx1250) — matrix operand order repaired and checked by reading, not compiled, not run on hardware
//
#include <hip/hip_runtime.h>

// ---------------------------------------------------------------------------
// sLSTM block for MI455X (gfx1250): bf16 WMMA GEMMs + TDM async tile feeds.
// B=8, S=2048, D=1024, H=1024.
// ---------------------------------------------------------------------------

#define BB 8
#define SS 2048
#define DD 1024
#define HH 1024

typedef __attribute__((ext_vector_type(16))) __bf16 v16bf;
typedef __attribute__((ext_vector_type(8)))  float  v8f;
typedef unsigned int u32x4 __attribute__((ext_vector_type(4)));
typedef int          i32x4 __attribute__((ext_vector_type(4)));
typedef int          i32x8 __attribute__((ext_vector_type(8)));

union Frag16 { uint4 u[2]; v16bf v; };

#if defined(__has_builtin)
#if __has_builtin(__builtin_amdgcn_tensor_load_to_lds) && \
    __has_builtin(__builtin_amdgcn_s_wait_tensorcnt)
#define HAVE_TDM 1
#endif
#endif
#ifndef HAVE_TDM
#define HAVE_TDM 0
#endif

__device__ __forceinline__ unsigned short f2bf(float f) {
    unsigned int u = __float_as_uint(f);
    unsigned int r = (u + 0x7FFFu + ((u >> 16) & 1u)) >> 16;
    return (unsigned short)r;
}
__device__ __forceinline__ float bf2f(unsigned short h) {
    return __uint_as_float(((unsigned int)h) << 16);
}

#if HAVE_TDM
// 2-D TDM tile load: tile_d0 x tile_d1 elements (2B each) from row-major
// tensor (row stride = stride0 elements) into LDS with HW padding.
__device__ __forceinline__ void tdm_load_2d(
    unsigned lds_addr, const void* gaddr,
    unsigned tile_d0, unsigned tile_d1,
    unsigned tensor_d0, unsigned tensor_d1,
    unsigned long long stride0,
    unsigned pad_interval, unsigned pad_amount)
{
    unsigned long long ga = (unsigned long long)(size_t)gaddr;
    u32x4 g0;
    g0.x = 1u;                                       // count=1, user desc
    g0.y = lds_addr;                                 // LDS byte address
    g0.z = (unsigned)(ga & 0xFFFFFFFFu);             // global_addr[31:0]
    g0.w = (unsigned)((ga >> 32) & 0x01FFFFFFu) | (2u << 30);   // type=2
    i32x8 g1;
    g1[0] = (int)((1u << 16)                 // data_size = 1 -> 2 bytes
                | (1u << 20)                 // pad_enable
                | (pad_interval << 22)
                | (pad_amount << 25));
    g1[1] = (int)((tensor_d0 & 0xFFFFu) << 16);
    g1[2] = (int)(((tensor_d0 >> 16) & 0xFFFFu) | ((tensor_d1 & 0xFFFFu) << 16));
    g1[3] = (int)(((tensor_d1 >> 16) & 0xFFFFu) | (tile_d0 << 16));
    g1[4] = (int)(tile_d1 & 0xFFFFu);                // tile_dim2 = 0
    g1[5] = (int)(stride0 & 0xFFFFFFFFull);
    g1[6] = (int)((stride0 >> 32) & 0xFFFFull);      // dim1_stride = 0
    g1[7] = 0;
    i32x4 z4 = {0, 0, 0, 0};
#if __clang_major__ >= 23
    i32x8 z8 = {0, 0, 0, 0, 0, 0, 0, 0};
    __builtin_amdgcn_tensor_load_to_lds(g0, g1, z4, z4, z8, 0);
#else
    __builtin_amdgcn_tensor_load_to_lds(g0, g1, z4, z4, 0);
#endif
}
#endif

// ---------------------------------------------------------------------------
// Transpose fp32 [R x C] (starting at row0) -> bf16 T[C][R]
// ---------------------------------------------------------------------------
__global__ __launch_bounds__(256) void transpose_bf16_k(
    const float* __restrict__ W, int row0, int R, int C,
    unsigned short* __restrict__ T)
{
    __shared__ float tile[32][33];
    int rb = blockIdx.y * 32, cb = blockIdx.x * 32;
    int tx = threadIdx.x & 31, ty = threadIdx.x >> 5;   // ty 0..7
    for (int i = 0; i < 32; i += 8)
        tile[ty + i][tx] = W[(size_t)(row0 + rb + ty + i) * C + cb + tx];
    __syncthreads();
    for (int i = 0; i < 32; i += 8)
        T[(size_t)(cb + ty + i) * R + rb + tx] = f2bf(tile[tx][ty + i]);
}

// ---------------------------------------------------------------------------
// Row-wise RMSNorm fp32 -> bf16 (one block per row, Dc = 1024)
// ---------------------------------------------------------------------------
__global__ __launch_bounds__(256) void rmsnorm_bf16_k(
    const float* __restrict__ x, const float* __restrict__ w,
    unsigned short* __restrict__ out, int Dc)
{
    __shared__ float red[256];
    int row = blockIdx.x;
    const float* xr = x + (size_t)row * Dc;
    float ss = 0.f;
    for (int j = threadIdx.x; j < Dc; j += 256) { float v = xr[j]; ss += v * v; }
    red[threadIdx.x] = ss; __syncthreads();
    for (int s = 128; s > 0; s >>= 1) {
        if (threadIdx.x < s) red[threadIdx.x] += red[threadIdx.x + s];
        __syncthreads();
    }
    float scale = rsqrtf(red[0] / (float)Dc + 1.1920929e-07f);
    for (int j = threadIdx.x; j < Dc; j += 256)
        out[(size_t)row * Dc + j] = f2bf(xr[j] * scale * w[j]);
}

// ---------------------------------------------------------------------------
// Generic bf16 WMMA GEMM: C = A[MxK] @ B[KxN] (+ epilogue), B given as BT[N][K]
// 128x128x32 tile per 256-thread workgroup (8 waves, each 32x64).
// TDM path: double-buffered LDS, DMA overlapped with WMMA, 1 barrier/iter.
// Modes:
//  0: outB[m][n]             = bf16(acc + bias[n])
//  1: outB[(m%S)*B+(m/S)][n] = bf16(acc + bias[n])          (bsd -> sbh)
//  2: outF[((m%B)*S+m/B)][n] = acc + bias[n] + resid[..]    (sbh -> bsd, +res)
//  3: outB[m][n]             = bf16(gelu(acc + bias[n]))    (exact erf gelu)
//  4: outF[m][n]             = acc + bias[n] + resid[m][n]
// ---------------------------------------------------------------------------
#define TM 128
#define TN 128
#define TK 32
#define APITCH 40   // 32 data + 8 pad bf16 per LDS row (80B, 16B aligned)

__global__ __launch_bounds__(256) void gemm_wmma_k(
    const unsigned short* __restrict__ A,   // [M][K] bf16
    const unsigned short* __restrict__ BT,  // [N][K] bf16
    const float* __restrict__ bias,
    const float* __restrict__ resid,
    float* __restrict__ outF,
    unsigned short* __restrict__ outB,
    int M, int N, int K, int S_, int Bb, int mode)
{
    __shared__ __align__(16) unsigned short As[2][TM * APITCH];
    __shared__ __align__(16) unsigned short Bs[2][TN * APITCH];

    const int tid  = threadIdx.x;
    const int lane = tid & 31;
    const int wave = tid >> 5;      // 0..7
    const int wrow = wave >> 1;     // 0..3  -> 32 rows
    const int wcol = wave & 1;      // 0..1  -> 64 cols
    const int m0 = blockIdx.y * TM;
    const int n0 = blockIdx.x * TN;

    v8f zero = {0.f, 0.f, 0.f, 0.f, 0.f, 0.f, 0.f, 0.f};
    v8f acc[2][4];
    for (int i = 0; i < 2; i++)
        for (int j = 0; j < 4; j++) acc[i][j] = zero;

    const int colL = lane & 15;
    const int kbA  = (lane >> 4) * 8;     // A frag K base: 0 / 8
    const int kbB  = (lane >> 4) * 8;     // B frag K base: 0 / 8

#if HAVE_TDM
    // ---- TDM double-buffered pipeline ----
    const unsigned ldsA[2] = { (unsigned)(size_t)&As[0][0], (unsigned)(size_t)&As[1][0] };
    const unsigned ldsB[2] = { (unsigned)(size_t)&Bs[0][0], (unsigned)(size_t)&Bs[1][0] };
    const int nk = K / TK;
    if (wave == 0) {
        tdm_load_2d(ldsA[0], A  + (size_t)m0 * K, TK, TM, K, M, K, 3, 3);
        tdm_load_2d(ldsB[0], BT + (size_t)n0 * K, TK, TN, K, N, K, 3, 3);
        __builtin_amdgcn_s_wait_tensorcnt(0);
    }
    __syncthreads();
    for (int ik = 0; ik < nk; ik++) {
        const int buf = ik & 1;
        if (ik + 1 < nk && wave == 0) {
            const int kn = (ik + 1) * TK;
            tdm_load_2d(ldsA[buf ^ 1], A  + (size_t)m0 * K + kn, TK, TM, K, M, K, 3, 3);
            tdm_load_2d(ldsB[buf ^ 1], BT + (size_t)n0 * K + kn, TK, TN, K, N, K, 3, 3);
        }
        const unsigned short* Ab = As[buf];
        const unsigned short* Bc = Bs[buf];
        Frag16 af[2];
        for (int i = 0; i < 2; i++) {
            const unsigned short* base = Ab + (wrow * 32 + i * 16 + colL) * APITCH;
            af[i].u[0] = *(const uint4*)(base + kbA);
            af[i].u[1] = *(const uint4*)(base + kbA + 16);
        }
        Frag16 bfrag[4];
        for (int j = 0; j < 4; j++) {
            const unsigned short* base = Bc + (wcol * 64 + j * 16 + colL) * APITCH + kbB;
            bfrag[j].u[0] = *(const uint4*)(base);
            bfrag[j].u[1] = *(const uint4*)(base + 16);
        }
        for (int i = 0; i < 2; i++)
            for (int j = 0; j < 4; j++)
                acc[i][j] = __builtin_amdgcn_wmma_f32_16x16x32_bf16(
                    false, af[i].v, false, bfrag[j].v, (short)0, acc[i][j],
                    false, false);
        if (ik + 1 < nk) {
            if (wave == 0) __builtin_amdgcn_s_wait_tensorcnt(0);
            __syncthreads();
        }
    }
#else
    // ---- fallback: manual staging, single buffer, 2 barriers/iter ----
    const int lrow   = tid >> 1;
    const int lchunk = (tid & 1) * 16;
    for (int k0 = 0; k0 < K; k0 += TK) {
        const uint4* gA = (const uint4*)(A + (size_t)(m0 + lrow) * K + k0 + lchunk);
        const uint4* gB = (const uint4*)(BT + (size_t)(n0 + lrow) * K + k0 + lchunk);
        uint4 a0 = gA[0], a1 = gA[1];
        uint4 b0 = gB[0], b1 = gB[1];
        if (k0 + TK < K) {
            __builtin_prefetch(A + (size_t)(m0 + lrow) * K + (k0 + TK) + lchunk, 0, 0);
            __builtin_prefetch(BT + (size_t)(n0 + lrow) * K + (k0 + TK) + lchunk, 0, 0);
        }
        __syncthreads();
        *(uint4*)(As[0] + lrow * APITCH + lchunk)     = a0;
        *(uint4*)(As[0] + lrow * APITCH + lchunk + 8) = a1;
        *(uint4*)(Bs[0] + lrow * APITCH + lchunk)     = b0;
        *(uint4*)(Bs[0] + lrow * APITCH + lchunk + 8) = b1;
        __syncthreads();
        Frag16 af[2];
        for (int i = 0; i < 2; i++) {
            const unsigned short* base = As[0] + (wrow * 32 + i * 16 + colL) * APITCH;
            af[i].u[0] = *(const uint4*)(base + kbA);
            af[i].u[1] = *(const uint4*)(base + kbA + 16);
        }
        Frag16 bfrag[4];
        for (int j = 0; j < 4; j++) {
            const unsigned short* base = Bs[0] + (wcol * 64 + j * 16 + colL) * APITCH + kbB;
            bfrag[j].u[0] = *(const uint4*)(base);
            bfrag[j].u[1] = *(const uint4*)(base + 16);
        }
        for (int i = 0; i < 2; i++)
            for (int j = 0; j < 4; j++)
                acc[i][j] = __builtin_amdgcn_wmma_f32_16x16x32_bf16(
                    false, af[i].v, false, bfrag[j].v, (short)0, acc[i][j],
                    false, false);
    }
#endif

    // epilogue: C layout — VGPR g: rows g / 8+g (lane-half), col = lane&15
    const int rowH = (lane >> 4) * 8;
    for (int i = 0; i < 2; i++) {
        for (int j = 0; j < 4; j++) {
            for (int g = 0; g < 8; g++) {
                int m = m0 + wrow * 32 + i * 16 + rowH + g;
                int n = n0 + wcol * 64 + j * 16 + colL;
                float v = acc[i][j][g] + bias[n];
                switch (mode) {
                case 0:
                    outB[(size_t)m * N + n] = f2bf(v);
                    break;
                case 1: {   // m = b*S + s  ->  pre[s][b][n]
                    int s = m % S_, b = m / S_;
                    outB[((size_t)s * Bb + b) * N + n] = f2bf(v);
                    break; }
                case 2: {   // m = s*B + b  ->  y[b][s][n] = v + resid
                    int s = m / Bb, b = m % Bb;
                    size_t o = ((size_t)b * S_ + s) * N + n;
                    outF[o] = v + resid[o];
                    break; }
                case 3: {   // exact gelu
                    float ge = 0.5f * v * (1.0f + erff(v * 0.70710678118654752f));
                    outB[(size_t)m * N + n] = f2bf(ge);
                    break; }
                case 4: {
                    size_t o = (size_t)m * N + n;
                    outF[o] = v + resid[o];
                    break; }
                }
            }
        }
    }
}

// ---------------------------------------------------------------------------
// Scan state init: c = 0, n = 1, h_bf16[16][1024] = 0 (rows 8..15 stay 0 pad)
// ---------------------------------------------------------------------------
__global__ __launch_bounds__(256) void scan_init_k(
    float* __restrict__ cst, float* __restrict__ nst,
    unsigned short* __restrict__ hb)
{
    int i = blockIdx.x * 256 + threadIdx.x;
    if (i < BB * HH) { cst[i] = 0.f; nst[i] = 1.f; }
    if (i < 16 * HH) hb[i] = 0;
}

// ---------------------------------------------------------------------------
// Recurrent gate GEMM: gout[gate][0..15][H] = h[16][H] @ Wh_gate
// WhT_g is [H][H] bf16 (K-contiguous, L2-resident). 32 WGs: 4 gates x 8 cols.
// h staged via TDM (pad 4 dwords per 256 dwords -> stride 520 dwords).
// ---------------------------------------------------------------------------
#define HPG 1040   // 1024 data + 16 pad bf16 per h row (2080B stride)

__global__ __launch_bounds__(256) void gate_gemm_k(
    const unsigned short* __restrict__ hb,     // [16][HH] bf16
    const unsigned short* __restrict__ whT_i,
    const unsigned short* __restrict__ whT_f,
    const unsigned short* __restrict__ whT_o,
    const unsigned short* __restrict__ whT_c,
    float* __restrict__ gout)                  // [4][16][HH]
{
    __shared__ __align__(16) unsigned short Hs[16 * HPG];
    const int tid = threadIdx.x, lane = tid & 31, wave = tid >> 5;
    const int gate = blockIdx.x >> 3;
    const int cg   = blockIdx.x & 7;
    const unsigned short* whT = (gate == 0) ? whT_i : (gate == 1) ? whT_f
                              : (gate == 2) ? whT_o : whT_c;

#if HAVE_TDM
    if (wave == 0) {
        // interval code 7 = 256 dwords, amount code 3 = 4 dwords (8 bf16)
        tdm_load_2d((unsigned)(size_t)Hs, hb, HH, 16, HH, 16, HH, 7, 3);
        __builtin_amdgcn_s_wait_tensorcnt(0);
    }
    __syncthreads();
#else
    for (int i = 0; i < 8; i++) {
        int e = (tid + i * 256) * 8;
        int r = e >> 10, col = e & 1023;
        *(uint4*)(Hs + r * HPG + col + ((col >> 9) * 8)) =
            *(const uint4*)(hb + r * HH + col);
    }
    __syncthreads();
#endif

    const int n0   = cg * 128 + wave * 16;
    const int colL = lane & 15;
    const int kbA  = (lane >> 4) * 8;
    const int kbB  = (lane >> 4) * 8;
    v8f acc = {0.f, 0.f, 0.f, 0.f, 0.f, 0.f, 0.f, 0.f};
    const unsigned short* brow = whT + (size_t)(n0 + colL) * HH;
    const unsigned short* arow = Hs + colL * HPG;

    for (int k0 = 0; k0 < HH; k0 += 32) {
        Frag16 a, b;
        int ka = k0 + kbA;
        a.u[0] = *(const uint4*)(arow + ka + ((ka >> 9) * 8));
        int ka2 = ka + 16;
        a.u[1] = *(const uint4*)(arow + ka2 + ((ka2 >> 9) * 8));
        const unsigned short* bbase = brow + k0 + kbB;
        b.u[0] = *(const uint4*)(bbase);
        b.u[1] = *(const uint4*)(bbase + 16);
        acc = __builtin_amdgcn_wmma_f32_16x16x32_bf16(
            false, a.v, false, b.v, (short)0, acc, false, false);
    }
    const int rowH = (lane >> 4) * 8;
    for (int g = 0; g < 8; g++) {
        int m = rowH + g;   // padded batch row 0..15 (only 0..7 consumed)
        gout[((size_t)gate * 16 + m) * HH + n0 + colL] = acc[g];
    }
}

// ---------------------------------------------------------------------------
// Cell update + layernorm for timestep s. One block per batch row b.
// ---------------------------------------------------------------------------
__global__ __launch_bounds__(256) void cell_update_k(
    const unsigned short* __restrict__ pre_i,
    const unsigned short* __restrict__ pre_f,
    const unsigned short* __restrict__ pre_o,
    const unsigned short* __restrict__ pre_c,
    const float* __restrict__ gout,
    const float* __restrict__ ln_g, const float* __restrict__ ln_b,
    float* __restrict__ cst, float* __restrict__ nst,
    unsigned short* __restrict__ hb,     // [16][HH] current h bf16
    unsigned short* __restrict__ hsb,    // [S][B][HH] all h, bf16
    int s)
{
    __shared__ float red[256];
    const int b = blockIdx.x, tid = threadIdx.x;
    const size_t preoff = ((size_t)s * BB + b) * HH;

    float vloc[4], oloc[4];
    float sum = 0.f;
    for (int k = 0; k < 4; k++) {
        int j = tid + k * 256;
        float pi = bf2f(pre_i[preoff + j]) + gout[((size_t)0 * 16 + b) * HH + j];
        float pf = bf2f(pre_f[preoff + j]) + gout[((size_t)1 * 16 + b) * HH + j];
        float po = bf2f(pre_o[preoff + j]) + gout[((size_t)2 * 16 + b) * HH + j];
        float pc = bf2f(pre_c[preoff + j]) + gout[((size_t)3 * 16 + b) * HH + j];
        float iv = expf(pi);
        float fv = expf(pf);
        float ov = 1.f / (1.f + expf(-po));
        float cc = tanhf(pc);
        float cprev = cst[b * HH + j], nprev = nst[b * HH + j];
        float nn = fv * nprev + iv;
        float cn = fv * cprev + iv * cc;
        cst[b * HH + j] = cn; nst[b * HH + j] = nn;
        float v = cn / (nn + 1e-6f);
        vloc[k] = v; oloc[k] = ov;
        sum += v;
    }
    red[tid] = sum; __syncthreads();
    for (int st = 128; st > 0; st >>= 1) {
        if (tid < st) red[tid] += red[tid + st];
        __syncthreads();
    }
    float mu = red[0] * (1.f / HH);
    __syncthreads();
    float sv = 0.f;
    for (int k = 0; k < 4; k++) { float d = vloc[k] - mu; sv += d * d; }
    red[tid] = sv; __syncthreads();
    for (int st = 128; st > 0; st >>= 1) {
        if (tid < st) red[tid] += red[tid + st];
        __syncthreads();
    }
    float rs = rsqrtf(red[0] * (1.f / HH) + 1e-5f);
    for (int k = 0; k < 4; k++) {
        int j = tid + k * 256;
        float hn = (vloc[k] - mu) * rs * ln_g[j] + ln_b[j];
        float h  = oloc[k] * tanhf(hn);
        unsigned short h16 = f2bf(h);
        hb[b * HH + j] = h16;
        hsb[((size_t)s * BB + b) * HH + j] = h16;
    }
}

// ---------------------------------------------------------------------------
// Host-side orchestration
// ---------------------------------------------------------------------------
extern "C" void kernel_launch(void* const* d_in, const int* in_sizes, int n_in,
                              void* d_out, int out_size, void* d_ws, size_t ws_size,
                              hipStream_t stream)
{
    (void)in_sizes; (void)n_in; (void)out_size; (void)ws_size;

    const float* x          = (const float*)d_in[0];
    const float* W_i        = (const float*)d_in[1];
    const float* b_i        = (const float*)d_in[2];
    const float* W_f        = (const float*)d_in[3];
    const float* b_f        = (const float*)d_in[4];
    const float* W_o        = (const float*)d_in[5];
    const float* b_o        = (const float*)d_in[6];
    const float* W_c        = (const float*)d_in[7];
    const float* b_c        = (const float*)d_in[8];
    const float* ln_g       = (const float*)d_in[9];
    const float* ln_b       = (const float*)d_in[10];
    const float* norm_w     = (const float*)d_in[11];
    const float* proj_in_W  = (const float*)d_in[12];
    const float* proj_in_b  = (const float*)d_in[13];
    const float* proj_out_W = (const float*)d_in[14];
    const float* proj_out_b = (const float*)d_in[15];
    const float* mlp_norm_w = (const float*)d_in[16];
    const float* mlp_W1     = (const float*)d_in[17];
    const float* mlp_b1     = (const float*)d_in[18];
    const float* mlp_W2     = (const float*)d_in[19];
    const float* mlp_b2     = (const float*)d_in[20];

    char* ws = (char*)d_ws;
    const size_t M  = (size_t)BB * SS;            // 16384
    const size_t WT = (size_t)DD * HH * 2;        // 2MB per [1024][1024] bf16

    size_t off = 0;
    auto take = [&](size_t bytes) { size_t r = off; off += (bytes + 255) & ~(size_t)255; return r; };

    size_t o_projinT  = take(WT);
    size_t o_wxT[4], o_whT[4];
    for (int g = 0; g < 4; g++) o_wxT[g] = take(WT);
    for (int g = 0; g < 4; g++) o_whT[g] = take(WT);
    size_t o_projoutT = take(WT);
    size_t o_w1T      = take((size_t)2 * DD * DD * 2);   // [2048][1024]
    size_t o_w2T      = take((size_t)2 * DD * DD * 2);   // [1024][2048]
    size_t o_xn       = take(M * DD * 2);                // reused as hs bf16
    size_t o_xp       = take(M * DD * 2);                // reused as mlp-norm bf16
    size_t o_pre      = take((size_t)4 * SS * BB * HH * 2); // reused as y(64MB)+t(64MB)
    size_t o_gout     = take((size_t)4 * 16 * HH * 4);
    size_t o_c        = take((size_t)BB * HH * 4);
    size_t o_n        = take((size_t)BB * HH * 4);
    size_t o_hb       = take((size_t)16 * HH * 2);

    unsigned short* projinT  = (unsigned short*)(ws + o_projinT);
    unsigned short* projoutT = (unsigned short*)(ws + o_projoutT);
    unsigned short* w1T      = (unsigned short*)(ws + o_w1T);
    unsigned short* w2T      = (unsigned short*)(ws + o_w2T);
    unsigned short* xn       = (unsigned short*)(ws + o_xn);
    unsigned short* hsb      = (unsigned short*)(ws + o_xn);       // reuse
    unsigned short* xp       = (unsigned short*)(ws + o_xp);
    unsigned short* mbf      = (unsigned short*)(ws + o_xp);       // reuse
    unsigned short* pre[4];
    for (int g = 0; g < 4; g++)
        pre[g] = (unsigned short*)(ws + o_pre + (size_t)g * SS * BB * HH * 2);
    float*          yv   = (float*)(ws + o_pre);                    // reuse (64MB)
    unsigned short* tbf  = (unsigned short*)(ws + o_pre + M * DD * 4); // reuse (64MB)
    float*          gout = (float*)(ws + o_gout);
    float*          cst  = (float*)(ws + o_c);
    float*          nst  = (float*)(ws + o_n);
    unsigned short* hb   = (unsigned short*)(ws + o_hb);
    float*          outp = (float*)d_out;

    // --- Stage 0: transpose all weights to bf16 [N][K] ---
    auto tr = [&](const float* W, int row0, int R, int C, unsigned short* T) {
        transpose_bf16_k<<<dim3(C / 32, R / 32), 256, 0, stream>>>(W, row0, R, C, T);
    };
    tr(proj_in_W, 0, DD, DD, projinT);
    const float* Wg[4] = {W_i, W_f, W_o, W_c};
    for (int g = 0; g < 4; g++) {
        tr(Wg[g], 0,  DD, HH, (unsigned short*)(ws + o_wxT[g]));   // Wx part
        tr(Wg[g], DD, HH, HH, (unsigned short*)(ws + o_whT[g]));   // Wh part
    }
    tr(proj_out_W, 0, HH, DD, projoutT);
    tr(mlp_W1, 0, DD, 2 * DD, w1T);
    tr(mlp_W2, 0, 2 * DD, DD, w2T);

    auto gemm = [&](const unsigned short* A, const unsigned short* BT,
                    const float* bias, const float* resid,
                    float* oF, unsigned short* oB, int Mm, int Nn, int Kk, int mode) {
        gemm_wmma_k<<<dim3(Nn / TN, Mm / TM), 256, 0, stream>>>(
            A, BT, bias, resid, oF, oB, Mm, Nn, Kk, SS, BB, mode);
    };

    // --- Stage 1: xn = rmsnorm(x, norm_w) -> bf16 ---
    rmsnorm_bf16_k<<<(int)M, 256, 0, stream>>>(x, norm_w, xn, DD);

    // --- Stage 2: xp = xn @ proj_in_W + b -> bf16 ---
    gemm(xn, projinT, proj_in_b, nullptr, nullptr, xp, (int)M, DD, DD, 0);

    // --- Stage 3: pre_g[s][b][h] = xp @ Wx_g + b_g -> bf16 (sbh layout) ---
    const float* bg[4] = {b_i, b_f, b_o, b_c};
    for (int g = 0; g < 4; g++)
        gemm(xp, (unsigned short*)(ws + o_wxT[g]), bg[g], nullptr,
             nullptr, pre[g], (int)M, HH, DD, 1);

    // --- Stage 4: recurrent scan ---
    scan_init_k<<<(16 * HH + 255) / 256, 256, 0, stream>>>(cst, nst, hb);
    for (int s = 0; s < SS; s++) {
        gate_gemm_k<<<32, 256, 0, stream>>>(
            hb,
            (unsigned short*)(ws + o_whT[0]), (unsigned short*)(ws + o_whT[1]),
            (unsigned short*)(ws + o_whT[2]), (unsigned short*)(ws + o_whT[3]),
            gout);
        cell_update_k<<<BB, 256, 0, stream>>>(
            pre[0], pre[1], pre[2], pre[3], gout, ln_g, ln_b,
            cst, nst, hb, hsb, s);
    }

    // --- Stage 5: y = hs @ proj_out_W + b + residual(x)  (sbh -> bsd) ---
    gemm(hsb, projoutT, proj_out_b, x, yv, nullptr, (int)M, DD, HH, 2);

    // --- Stage 6: m = rmsnorm(y, mlp_norm_w) -> bf16 ---
    rmsnorm_bf16_k<<<(int)M, 256, 0, stream>>>(yv, mlp_norm_w, mbf, DD);

    // --- Stage 7: t = gelu(m @ W1 + b1) -> bf16 ---
    gemm(mbf, w1T, mlp_b1, nullptr, nullptr, tbf, (int)M, 2 * DD, DD, 3);

    // --- Stage 8: out = y + (t @ W2 + b2) ---
    gemm(tbf, w2T, mlp_b2, yv, outp, nullptr, (int)M, DD, 2 * DD, 4);
}
